// ModelClass_61967788147287
// MI455X (gfx1250) — matrix operand order repaired and checked by reading, not compiled, not run on hardware
//
#include <hip/hip_runtime.h>
#include <hip/hip_bf16.h>

// ---------------------------------------------------------------------------
// GIN tracking-GNN for MI455X (gfx1250): fused 3-layer MLP via f16 WMMA
// (f32 accumulate). Weights staged per-k-slice into LDS with
// GLOBAL_LOAD_ASYNC_TO_LDS_B128 (ASYNCcnt) when available.
// ---------------------------------------------------------------------------

#define NLAYERS 5
#define FEAT    80     // D + S
#define KP1     96     // FEAT padded to multiple of 32 for WMMA K
#define HID     256
#define DYN     64
#define ROWS    64     // node rows per block in the MLP kernel

typedef _Float16 h16;
typedef __attribute__((ext_vector_type(16))) _Float16 v16h;
typedef __attribute__((ext_vector_type(8)))  float    v8f;

union FragU { v16h v; uint4 q[2]; };

#if defined(__gfx1250__) && __has_builtin(__builtin_amdgcn_global_load_async_to_lds_b128)
#define USE_ASYNC_LDS 1
#else
#define USE_ASYNC_LDS 0
#endif

typedef int v4i_ __attribute__((vector_size(16)));
typedef __attribute__((address_space(1))) v4i_ gas_v4i;   // global (AS1)
typedef __attribute__((address_space(3))) v4i_ las_v4i;   // LDS (AS3)

// A fragment (16x32 f16) from row-major LDS tile.
// Lane L: row m = L%16; halves 0..7  = K[(L<16?0:8) + 0..7]
//                       halves 8..15 = K[16 + (L<16?0:8) + 0..7]
__device__ __forceinline__ v16h load_a(const h16* base, int stride, int m,
                                       int kbase, int hi) {
  FragU u;
  const h16* p = base + m * stride + kbase + hi * 8;
  u.q[0] = *(const uint4*)(p);
  u.q[1] = *(const uint4*)(p + 16);
  return u.v;
}

// B fragment (32x16 f16) from an LDS k-slice laid out [n][32] halfs.
// Lane L: col n = L%16; halves 0..7 = K[(L<16?0:8) + 0..7], 8..15 = K[16 + (L<16?0:8) + 0..7].
__device__ __forceinline__ v16h load_b_lds(const h16* bl, int n, int hi) {
  FragU u;
  const h16* p = bl + n * 32 + hi * 8;
  u.q[0] = *(const uint4*)(p);
  u.q[1] = *(const uint4*)(p + 16);
  return u.v;
}

// Cooperatively copy W[n][kt*32 .. kt*32+32) for n in [0,nrows) into
// bl[n*32 ..], 16-byte chunks, 128 threads. Async-to-LDS when available.
__device__ __forceinline__ void stage_b_slice(h16* bl, const h16* W, int kp,
                                              int nrows, int kt, int tid) {
  const int chunks = nrows * 4;                  // 16B chunks
  for (int id = tid; id < chunks; id += 128) {
    int n = id >> 2, part = id & 3;
    const h16* gp = W + (size_t)n * kp + kt * 32 + part * 8;
    h16* lp = bl + n * 32 + part * 8;
#if USE_ASYNC_LDS
    __builtin_amdgcn_global_load_async_to_lds_b128((gas_v4i*)gp, (las_v4i*)lp,
                                                   0, 0);
#else
    *(uint4*)lp = *(const uint4*)gp;
#endif
  }
#if USE_ASYNC_LDS
#if __has_builtin(__builtin_amdgcn_s_wait_asynccnt)
  __builtin_amdgcn_s_wait_asynccnt(0);
#else
  asm volatile("s_wait_asynccnt 0x0" ::: "memory");
#endif
#endif
}

// ---------------------------------------------------------------------------
// Fused MLP: zbuf (P x 80 f32) -> relu(L1) -> relu(L2) -> relu(L3) -> xout
// Block: 128 threads (4 waves). Each block: 64 nodes. Wave w: m-tile w.
// ---------------------------------------------------------------------------
__global__ void __launch_bounds__(128)
mlp_kernel(const float* __restrict__ zbuf, float* __restrict__ xout,
           const h16* __restrict__ Wt1, const h16* __restrict__ Wt2,
           const h16* __restrict__ Wt3, const float* __restrict__ b1,
           const float* __restrict__ b2, const float* __restrict__ b3)
{
  __shared__ h16 zl[ROWS * KP1];   // 12 KB input tile
  __shared__ h16 hl[ROWS * HID];   // 32 KB hidden (h1 then h2)
  __shared__ h16 bl[HID * 32];     // 16 KB weight k-slice [n][32]

  const int tid  = threadIdx.x;
  const int row0 = blockIdx.x * ROWS;

  // stage 0: stage input tile into LDS as f16, zero-pad K 80->96
  for (int i = tid; i < ROWS * KP1; i += 128) {
    int r = i / KP1, c = i - r * KP1;
    float v = (c < FEAT) ? zbuf[(row0 + r) * FEAT + c] : 0.f;
    zl[i] = (h16)v;
  }

  const int w    = tid >> 5;
  const int lane = tid & 31;
  const int lh   = lane & 15;
  const int hi   = lane >> 4;
  const int m    = w * 16 + lh;      // A-fragment row within tile

  v8f acc[16];
  v16h bf[16];

  // ---- stage 1: (64x96) @ (96x256) ----
  #pragma unroll
  for (int nt = 0; nt < 16; ++nt)
    #pragma unroll
    for (int r = 0; r < 8; ++r) acc[nt][r] = 0.f;

  for (int kt = 0; kt < 3; ++kt) {
    __syncthreads();                       // prev slice consumed / zl staged
    stage_b_slice(bl, Wt1, KP1, HID, kt, tid);
    __syncthreads();
    v16h a = load_a(zl, KP1, m, kt * 32, hi);
    #pragma unroll
    for (int nt = 0; nt < 16; ++nt) bf[nt] = load_b_lds(bl, nt * 16 + lh, hi);
    #pragma unroll
    for (int nt = 0; nt < 16; ++nt)
      acc[nt] = __builtin_amdgcn_wmma_f32_16x16x32_f16(
          false, a, false, bf[nt], (short)0, acc[nt], false, false);
  }
  #pragma unroll
  for (int nt = 0; nt < 16; ++nt) {
    int n = nt * 16 + lh;
    float bb = b1[n];
    #pragma unroll
    for (int r = 0; r < 8; ++r) {
      float v = acc[nt][r] + bb;
      v = v > 0.f ? v : 0.f;
      hl[(w * 16 + r + hi * 8) * HID + n] = (h16)v;
    }
  }

  // ---- stage 2: (64x256) @ (256x256) ----
  #pragma unroll
  for (int nt = 0; nt < 16; ++nt)
    #pragma unroll
    for (int r = 0; r < 8; ++r) acc[nt][r] = 0.f;

  for (int kt = 0; kt < 8; ++kt) {
    __syncthreads();                       // h1 written / prev slice consumed
    stage_b_slice(bl, Wt2, HID, HID, kt, tid);
    __syncthreads();
    v16h a = load_a(hl, HID, m, kt * 32, hi);
    #pragma unroll
    for (int nt = 0; nt < 16; ++nt) bf[nt] = load_b_lds(bl, nt * 16 + lh, hi);
    #pragma unroll
    for (int nt = 0; nt < 16; ++nt)
      acc[nt] = __builtin_amdgcn_wmma_f32_16x16x32_f16(
          false, a, false, bf[nt], (short)0, acc[nt], false, false);
  }
  __syncthreads();   // all reads of h1 done before overwrite
  #pragma unroll
  for (int nt = 0; nt < 16; ++nt) {
    int n = nt * 16 + lh;
    float bb = b2[n];
    #pragma unroll
    for (int r = 0; r < 8; ++r) {
      float v = acc[nt][r] + bb;
      v = v > 0.f ? v : 0.f;
      hl[(w * 16 + r + hi * 8) * HID + n] = (h16)v;
    }
  }

  // ---- stage 3: (64x256) @ (256x64) -> global x ----
  v8f a3[4];
  #pragma unroll
  for (int nt = 0; nt < 4; ++nt)
    #pragma unroll
    for (int r = 0; r < 8; ++r) a3[nt][r] = 0.f;

  for (int kt = 0; kt < 8; ++kt) {
    __syncthreads();                       // h2 written / prev slice consumed
    stage_b_slice(bl, Wt3, HID, DYN, kt, tid);
    __syncthreads();
    v16h a = load_a(hl, HID, m, kt * 32, hi);
    #pragma unroll
    for (int nt = 0; nt < 4; ++nt) bf[nt] = load_b_lds(bl, nt * 16 + lh, hi);
    #pragma unroll
    for (int nt = 0; nt < 4; ++nt)
      a3[nt] = __builtin_amdgcn_wmma_f32_16x16x32_f16(
          false, a, false, bf[nt], (short)0, a3[nt], false, false);
  }
  #pragma unroll
  for (int nt = 0; nt < 4; ++nt) {
    int n = nt * 16 + lh;
    float bb = b3[n];
    #pragma unroll
    for (int r = 0; r < 8; ++r) {
      float v = a3[nt][r] + bb;
      v = v > 0.f ? v : 0.f;
      xout[(size_t)(row0 + w * 16 + r + hi * 8) * DYN + n] = v;
    }
  }
}

// ---------------------------------------------------------------------------
// zbuf[p][f] = scale * concat(x, static)[node_base + p][f]
// ---------------------------------------------------------------------------
__global__ void init_z_kernel(float* __restrict__ z, const float* __restrict__ x,
                              const float* __restrict__ st, int node_base,
                              const float* __restrict__ eps, int eps_idx, int P)
{
  int t = blockIdx.x * blockDim.x + threadIdx.x;
  if (t >= P * FEAT) return;
  int p = t / FEAT, f = t - p * FEAT;
  int n = node_base + p;
  float v = (f < DYN) ? x[(size_t)n * DYN + f] : st[(size_t)n * 16 + (f - DYN)];
  float s = (eps_idx >= 0) ? (1.f + eps[eps_idx]) : 1.f;
  z[t] = v * s;
}

// ---------------------------------------------------------------------------
// zbuf[dst[e]][f] += concat(x, static)[src_base + src[e]][f]
// ---------------------------------------------------------------------------
__global__ void scatter_kernel(float* __restrict__ z, const float* __restrict__ x,
                               const float* __restrict__ st,
                               const int* __restrict__ src,
                               const int* __restrict__ dst, int src_base, int nE)
{
  int t = blockIdx.x * blockDim.x + threadIdx.x;
  if (t >= nE * FEAT) return;
  int e = t / FEAT, f = t - e * FEAT;
  int sn = src_base + src[e];
  float v = (f < DYN) ? x[(size_t)sn * DYN + f] : st[(size_t)sn * 16 + (f - DYN)];
  atomicAdd(&z[(size_t)dst[e] * FEAT + f], v);
}

__global__ void xinit_kernel(float* __restrict__ x, const float* __restrict__ x0,
                             int N)
{
  int t = blockIdx.x * blockDim.x + threadIdx.x;
  if (t >= N * DYN) return;
  int n = t >> 6, c = t & 63;
  x[t] = (c == 0) ? x0[n] : 0.f;
}

// dst[i][n][kp] = f16(src[i][k][n]), zero-padded k >= K
__global__ void wprep_kernel(h16* __restrict__ dst, const float* __restrict__ src,
                             int K, int Nn, int Kp, int count)
{
  int t = blockIdx.x * blockDim.x + threadIdx.x;
  int per = Nn * Kp;
  if (t >= count * per) return;
  int i = t / per, rem = t - i * per;
  int n = rem / Kp, k = rem - n * Kp;
  float v = (k < K) ? src[((size_t)i * K + k) * Nn + n] : 0.f;
  dst[t] = (h16)v;
}

__global__ void pzero_kernel(float* __restrict__ p, int n) {
  int t = blockIdx.x * blockDim.x + threadIdx.x;
  if (t < n) p[t] = 0.f;
}

__global__ void pscat_kernel(float* __restrict__ pool, const float* __restrict__ x,
                             const int* __restrict__ bidx, int N)
{
  int t = blockIdx.x * blockDim.x + threadIdx.x;
  if (t >= N * DYN) return;
  int n = t >> 6, c = t & 63;
  atomicAdd(&pool[(size_t)bidx[n] * DYN + c], x[t]);
}

__global__ void pfinal_kernel(float* __restrict__ out, const float* __restrict__ pool,
                              const float* __restrict__ lw, const float* __restrict__ lb,
                              int NG)
{
  int g = blockIdx.x * blockDim.x + threadIdx.x;
  if (g >= NG) return;
  float s = 0.f;
  for (int c = 0; c < DYN; ++c) s += pool[(size_t)g * DYN + c] * lw[c];
  s += lb[0];
  out[g] = s > 0.f ? s : 0.f;
}

// ---------------------------------------------------------------------------
extern "C" void kernel_launch(void* const* d_in, const int* in_sizes, int n_in,
                              void* d_out, int out_size, void* d_ws, size_t ws_size,
                              hipStream_t stream)
{
  const float* x0   = (const float*)d_in[0];
  const float* st   = (const float*)d_in[1];
  const float* W1   = (const float*)d_in[2];
  const float* b1   = (const float*)d_in[3];
  const float* W2   = (const float*)d_in[4];
  const float* b2   = (const float*)d_in[5];
  const float* W3   = (const float*)d_in[6];
  const float* b3   = (const float*)d_in[7];
  const float* eps  = (const float*)d_in[8];
  const float* lw   = (const float*)d_in[9];
  const float* lb   = (const float*)d_in[10];
  const int* isrc   = (const int*)d_in[11];
  const int* idst   = (const int*)d_in[12];
  const int* fsrc   = (const int*)d_in[13];
  const int* fdst   = (const int*)d_in[14];
  const int* bsrc   = (const int*)d_in[15];
  const int* bdst   = (const int*)d_in[16];
  const int* bidx   = (const int*)d_in[17];

  const int N  = in_sizes[0];             // 98304
  const int P  = N / (NLAYERS + 1);       // 16384
  const int E  = in_sizes[11] / NLAYERS;  // 262144
  const int NG = out_size;                // 64

  // workspace layout
  float* xbuf = (float*)d_ws;                       // N*64 f32
  float* zbuf = xbuf + (size_t)N * DYN;             // P*80 f32
  float* pool = zbuf + (size_t)P * FEAT;            // NG*64 f32
  h16*   Wt1  = (h16*)(pool + (size_t)NG * DYN);    // 4 * 256*96 f16
  h16*   Wt2  = Wt1 + 4 * (HID * KP1);              // 4 * 256*256 f16
  h16*   Wt3  = Wt2 + 4 * (HID * HID);              // 4 * 64*256 f16
  (void)ws_size; (void)n_in;

  const int T = 256;
  // weight transpose/convert (tiny; weights then live in L2 for the whole run)
  wprep_kernel<<<(4 * HID * KP1 + T - 1) / T, T, 0, stream>>>(Wt1, W1, FEAT, HID, KP1, 4);
  wprep_kernel<<<(4 * HID * HID + T - 1) / T, T, 0, stream>>>(Wt2, W2, HID, HID, HID, 4);
  wprep_kernel<<<(4 * DYN * HID + T - 1) / T, T, 0, stream>>>(Wt3, W3, HID, DYN, HID, 4);
  xinit_kernel<<<(N * DYN + T - 1) / T, T, 0, stream>>>(xbuf, x0, N);

  auto initz = [&](int node_base, int eps_idx) {
    init_z_kernel<<<(P * FEAT + T - 1) / T, T, 0, stream>>>(
        zbuf, xbuf, st, node_base, eps, eps_idx, P);
  };
  auto scat = [&](const int* s, const int* d, int src_base) {
    scatter_kernel<<<(E * FEAT + T - 1) / T, T, 0, stream>>>(
        zbuf, xbuf, st, s, d, src_base, E);
  };
  auto mlp = [&](int i, int target_base) {
    mlp_kernel<<<P / ROWS, 128, 0, stream>>>(
        zbuf, xbuf + (size_t)target_base * DYN,
        Wt1 + (size_t)i * (HID * KP1), Wt2 + (size_t)i * (HID * HID),
        Wt3 + (size_t)i * (DYN * HID),
        b1 + i * HID, b2 + i * HID, b3 + i * DYN);
  };

  // forward sweep
  for (int il = 0; il < NLAYERS; ++il) {
    int s0 = il * P;
    initz(s0, 0);                                        // inner conv, eps[0]
    scat(isrc + (size_t)il * E, idst + (size_t)il * E, s0);
    mlp(0, s0);
    if (il == NLAYERS - 1) continue;
    initz(s0 + P, 1);                                    // forward conv, eps[1]
    scat(fsrc + (size_t)il * E, fdst + (size_t)il * E, s0);
    mlp(1, s0 + P);
    initz(s0 + P, -1);                                   // node_dnn
    mlp(3, s0 + P);
  }
  // backward sweep
  for (int il = NLAYERS - 1; il >= 1; --il) {
    int s0 = (il - 1) * P;
    initz(s0, 2);                                        // backward conv, eps[2]
    scat(bsrc + (size_t)(il - 1) * E, bdst + (size_t)(il - 1) * E, s0 + P);
    mlp(2, s0);
    initz(s0, 0);                                        // inner conv
    scat(isrc + (size_t)(il - 1) * E, idst + (size_t)(il - 1) * E, s0);
    mlp(0, s0);
    initz(s0, -1);                                       // node_dnn
    mlp(3, s0);
  }

  // pooling + head
  pzero_kernel<<<(NG * DYN + T - 1) / T, T, 0, stream>>>(pool, NG * DYN);
  pscat_kernel<<<(N * DYN + T - 1) / T, T, 0, stream>>>(pool, xbuf, bidx, N);
  pfinal_kernel<<<1, NG, 0, stream>>>((float*)d_out, pool, lw, lb, NG);
}
